// NLLvMF_60868276519354
// MI455X (gfx1250) — compile-verified, matrix operand order checked by reading, not run on hardware
//
#include <hip/hip_runtime.h>
#include <math.h>

// CDNA5 / gfx1250: wave32, WMMA 16x16x32 f16->f32
typedef __attribute__((ext_vector_type(16))) _Float16 v16h;
typedef __attribute__((ext_vector_type(8)))  float    v8f;

#define D_FEAT          50
#define V_ORD           24.0f          // m/2 - 1 with m = 50
#define N_SERIES        200
#define HALF_M_LOG_2PI  45.946926660233635f   // 25 * log(2*pi)
#define THREADS         256
#define ROWS_PER_BLOCK  256            // one row per thread for the series phase

// Stage 1: per-256-row block -> partial sum of log_cmk
__global__ __launch_bounds__(THREADS)
void nllvmf_stage1(const float* __restrict__ input, float* __restrict__ partial)
{
    __shared__ float kappa_lds[ROWS_PER_BLOCK];
    __shared__ float red[THREADS];

    const int tid  = threadIdx.x;
    const int lane = tid & 31;
    const int wave = tid >> 5;
    const long rowBlock = (long)blockIdx.x * ROWS_PER_BLOCK;

    // All-ones B matrix: D[m,n] = sum_k A[m,k] -> every D column is kappa.
    v16h ones;
#pragma unroll
    for (int i = 0; i < 16; ++i) ones[i] = (_Float16)1.0f;

    // Each wave computes kappa for 32 rows via two 16-row WMMA tiles.
#pragma unroll
    for (int t = 0; t < 2; ++t) {
        const int  rowLocal = wave * 32 + t * 16 + (lane & 15);
        const long row      = rowBlock + rowLocal;
        const float* rp     = input + row * (long)D_FEAT;
        const int  half     = lane >> 4;   // which 16-col slice this lane packs

        v16h hi0, lo0, hi1, lo1;

        // chunk0: columns [half*16, half*16+16) -- always in range (< 32 < 50)
        {
            const int cb = half * 16;
#pragma unroll
            for (int k = 0; k < 8; ++k) {
                const float2 xy = *(const float2*)(rp + cb + 2 * k);
                const float a0 = fabsf(xy.x);
                const float a1 = fabsf(xy.y);
                const _Float16 h0 = (_Float16)a0;
                const _Float16 h1 = (_Float16)a1;
                hi0[2 * k]     = h0;
                hi0[2 * k + 1] = h1;
                lo0[2 * k]     = (_Float16)(a0 - (float)h0);  // error-compensation term
                lo0[2 * k + 1] = (_Float16)(a1 - (float)h1);
            }
        }
        // chunk1: columns [32+half*16, 32+half*16+16) -- tail padded with zeros
        {
            const int cb = 32 + half * 16;
#pragma unroll
            for (int k = 0; k < 16; ++k) {
                const int col = cb + k;
                const float a = (col < D_FEAT) ? fabsf(rp[col]) : 0.0f;
                const _Float16 h = (_Float16)a;
                hi1[k] = h;
                lo1[k] = (_Float16)(a - (float)h);
            }
        }

        // kappa[m] accumulated in f32: hi + lo split keeps ~f32 accuracy.
        v8f c = {};
        c = __builtin_amdgcn_wmma_f32_16x16x32_f16(false, hi0, false, ones, (short)0, c, false, false);
        c = __builtin_amdgcn_wmma_f32_16x16x32_f16(false, lo0, false, ones, (short)0, c, false, false);
        c = __builtin_amdgcn_wmma_f32_16x16x32_f16(false, hi1, false, ones, (short)0, c, false, false);
        c = __builtin_amdgcn_wmma_f32_16x16x32_f16(false, lo1, false, ones, (short)0, c, false, false);

        // D layout (32-bit C/D 16x16): lanes 0-15 VGPR r = M=r; lanes 16-31 VGPR r = M=8+r.
        // Every column is identical, so lane 0 / lane 16 carry all 16 kappas.
        if (lane == 0) {
#pragma unroll
            for (int r = 0; r < 8; ++r) kappa_lds[wave * 32 + t * 16 + r] = c[r];
        }
        if (lane == 16) {
#pragma unroll
            for (int r = 0; r < 8; ++r) kappa_lds[wave * 32 + t * 16 + 8 + r] = c[r];
        }
    }

    __syncthreads();

    // ---- series phase: one row per thread ----
    // log I_v(x) = v*lx + logsumexp_j(2j*lx - lgamma(j+1) - lgamma(v+j+1)), lx = log(x/2)
    // Terms are concave in j; peak at j* where (j+1)(v+j+1) ~= (x/2)^2.
    // Sum in linear space relative to the peak (ratio recurrences): no exp needed.
    const float x  = kappa_lds[tid];
    const float lx = logf(0.5f * x);
    const float q  = 0.25f * x * x;                       // (x/2)^2

    const float js_f = 0.5f * (-V_ORD + sqrtf(V_ORD * V_ORD + 4.0f * q));
    int js = (int)js_f;
    js = js < 0 ? 0 : (js > N_SERIES - 1 ? N_SERIES - 1 : js);

    const float jsf    = (float)js;
    const float t_peak = 2.0f * jsf * lx - lgammaf(jsf + 1.0f) - lgammaf(V_ORD + jsf + 1.0f);

    float sum  = 1.0f;   // peak term, normalized
    float term = 1.0f;
    for (int j = js; j < N_SERIES - 1; ++j) {             // ascending tail
        const float jp1 = (float)(j + 1);
        term *= q / (jp1 * (V_ORD + jp1));
        sum  += term;
    }
    term = 1.0f;
    for (int j = js; j >= 1; --j) {                       // descending tail
        const float jf = (float)j;
        term *= (jf * (V_ORD + jf)) / q;
        sum  += term;
    }

    const float log_iv  = V_ORD * lx + t_peak + logf(sum);
    const float log_cmk = V_ORD * logf(x) - log_iv - HALF_M_LOG_2PI;

    // ---- block reduction ----
    red[tid] = log_cmk;
    __syncthreads();
#pragma unroll
    for (int s = THREADS / 2; s > 0; s >>= 1) {
        if (tid < s) red[tid] += red[tid + s];
        __syncthreads();
    }
    if (tid == 0) partial[blockIdx.x] = red[0];
}

// Stage 2: reduce the per-block partials, write -mean
__global__ __launch_bounds__(256)
void nllvmf_stage2(const float* __restrict__ partial, int nPartials,
                   float* __restrict__ out, float invB)
{
    __shared__ float red[256];
    float s = 0.0f;
    for (int i = threadIdx.x; i < nPartials; i += 256) s += partial[i];
    red[threadIdx.x] = s;
    __syncthreads();
#pragma unroll
    for (int st = 128; st > 0; st >>= 1) {
        if (threadIdx.x < st) red[threadIdx.x] += red[threadIdx.x + st];
        __syncthreads();
    }
    if (threadIdx.x == 0) out[0] = -red[0] * invB;
}

extern "C" void kernel_launch(void* const* d_in, const int* in_sizes, int n_in,
                              void* d_out, int out_size, void* d_ws, size_t ws_size,
                              hipStream_t stream)
{
    const float* input = (const float*)d_in[0];
    // d_in[1] ("target") is unused by the reference computation.
    const int B = in_sizes[0] / D_FEAT;            // 524288
    const int nBlocks = B / ROWS_PER_BLOCK;        // 2048 (B divisible by 256)

    float* partial = (float*)d_ws;                 // 2048 floats of scratch

    nllvmf_stage1<<<nBlocks, THREADS, 0, stream>>>(input, partial);
    nllvmf_stage2<<<1, 256, 0, stream>>>(partial, nBlocks, (float*)d_out,
                                         1.0f / (float)B);
}
